// SelfAttention_19370302505359
// MI455X (gfx1250) — matrix operand order repaired and checked by reading, not compiled, not run on hardware
//
#include <hip/hip_runtime.h>

// ---------------------------------------------------------------------------
// SelfAttention, B=16 N=4096 D=128, for MI455X (gfx1250, wave32).
//
// Exact decomposition:  scores[b,i,j] = G_ij + x_bi*u_j + x_bj*v_i + c0*x_bi*x_bj
// with G = A @ B^T batch-independent (A_i = q_b+pe_i, B_j = k_b+pe_j).
// Only 2*N*N*D = 4.3 GFLOP of matrix work (WMMA f16, negligible); dominant
// cost is the mandatory B*N*N = 268M exp stream -> one v_exp_f32 per element
// (scores are O(0.5), so softmax needs no max-subtraction; scale*log2e is
// pre-folded into all coefficients so exp2 is a bare v_exp_f32).
//
// Data movement: B tiles (4 KB) staged into LDS by the Tensor Data Mover
// (tensor_load_to_lds, double-buffered, s_wait_tensorcnt + barrier); all 8
// waves per WGP read fragments from LDS (one clause of 8 ds_load_b128 per
// j-tile, then 4 back-to-back WMMAs) while the next tile DMAs in.
// Workspace: Ah(1MB) + Bh(1MB) + u(16KB) + v(16KB) + c0 = ~2.13 MB.
// ---------------------------------------------------------------------------

typedef _Float16 h4    __attribute__((ext_vector_type(4)));
typedef _Float16 h8    __attribute__((ext_vector_type(8)));
typedef _Float16 v16h  __attribute__((ext_vector_type(16)));
typedef float    v8f   __attribute__((ext_vector_type(8)));
typedef unsigned u32x4 __attribute__((ext_vector_type(4)));
typedef unsigned u32x8 __attribute__((ext_vector_type(8)));

#define NN 4096
#define ND 128
#define NJT (NN / 16)
#define SCALE 0.08838834764831845f                /* 1/sqrt(128) */
#define K2 (SCALE * 1.4426950408889634f)          /* fold log2(e) for v_exp_f32 */

struct F4 { float f[4]; };

// Generic LDS pointer = {SHARED_BASE, offset[31:0]} (ISA §10.2): low 32 bits
// are the hardware LDS byte address used by the TDM descriptor.
static __device__ __forceinline__ unsigned lds_offset(const void* p) {
    return (unsigned)(unsigned long long)p;
}

// Issue one TDM transfer: 16 rows x 128 f16 (4 KB) Global -> LDS.
// D# per ISA §8: group0 = {count=1 | lds_addr | global_addr | type=2},
// group1 = {data_size=2B, tensor_dim0=128, tensor_dim1=4096, tile=128x16,
// dim0_stride=128}.  Inline asm (not the builtin) for toolchain portability.
static __device__ __forceinline__ void tdm_load_tile(const _Float16* gsrc,
                                                     unsigned lds_addr) {
    unsigned long long ga = (unsigned long long)gsrc;
    u32x4 g0;
    g0.x = 1u;                                   // count=1, user mode, no gather
    g0.y = lds_addr;                             // lds_addr (bytes)
    g0.z = (unsigned)ga;                         // global_addr[31:0]
    g0.w = (unsigned)(ga >> 32) | 0x80000000u;   // global_addr[56:32] | type=2
    u32x8 g1;
    g1[0] = 1u << 16;                            // data_size=1 -> 2 bytes/elem
    g1[1] = 128u << 16;                          // tensor_dim0[15:0]=128 @bits63:48
    g1[2] = (unsigned)NN << 16;                  // tensor_dim1[15:0]=4096 @bits95:80
    g1[3] = 128u << 16;                          // tile_dim0=128 @bits127:112
    g1[4] = 16u;                                 // tile_dim1=16, tile_dim2=0
    g1[5] = 128u;                                // tensor_dim0_stride=128 elems
    g1[6] = 0u;
    g1[7] = 0u;
    asm volatile("tensor_load_to_lds %0, %1" :: "s"(g0), "s"(g1) : "memory");
}

// ---------------------------------------------------------------------------
// Phase 1: Ah[i][d]=(f16)(q_b[d]+pe[i][d]); Bh[j][d]=(f16)(k_b[d]+pe[j][d]);
//          u[j]=qw.B_j ; v[i]=kw.A_i ; c0=qw.kw    (one wave per row)
// ---------------------------------------------------------------------------
__global__ void __launch_bounds__(256) sa_prep_kernel(
    const float* __restrict__ q_w, const float* __restrict__ q_b,
    const float* __restrict__ k_w, const float* __restrict__ k_b,
    const float* __restrict__ pe,
    _Float16* __restrict__ Ah, _Float16* __restrict__ Bh,
    float* __restrict__ u, float* __restrict__ v, float* __restrict__ c0)
{
    const int wave = threadIdx.x >> 5;
    const int lane = threadIdx.x & 31;
    const int row  = blockIdx.x * 8 + wave;     // 512 blocks * 8 waves = 4096
    const int d0   = lane * 4;                  // 32 lanes * 4 dims = 128

    const F4 pe4 = *(const F4*)(pe + row * ND + d0);
    const F4 qb4 = *(const F4*)(q_b + d0);
    const F4 kb4 = *(const F4*)(k_b + d0);
    const F4 qw4 = *(const F4*)(q_w + d0);
    const F4 kw4 = *(const F4*)(k_w + d0);

    h4 av, bv;
    float us = 0.f, vs = 0.f;
#pragma unroll
    for (int t = 0; t < 4; ++t) {
        float a = qb4.f[t] + pe4.f[t];
        float b = kb4.f[t] + pe4.f[t];
        av[t] = (_Float16)a;
        bv[t] = (_Float16)b;
        us += qw4.f[t] * b;
        vs += kw4.f[t] * a;
    }
    *(h4*)(Ah + row * ND + d0) = av;
    *(h4*)(Bh + row * ND + d0) = bv;

#pragma unroll
    for (int off = 16; off >= 1; off >>= 1) {
        us += __shfl_xor(us, off, 32);
        vs += __shfl_xor(vs, off, 32);
    }
    if (lane == 0) { u[row] = us; v[row] = vs; }

    if (blockIdx.x == 0 && wave == 0) {
        float cs = 0.f;
#pragma unroll
        for (int t = 0; t < 4; ++t) cs += qw4.f[t] * kw4.f[t];
#pragma unroll
        for (int off = 16; off >= 1; off >>= 1) cs += __shfl_xor(cs, off, 32);
        if (lane == 0) *c0 = cs;
    }
}

// ---------------------------------------------------------------------------
// Phase 2: TDM-staged WMMA G tiles + fused softmax accumulation.
// Grid: 256 blocks (one 16-row i-tile) x 8 waves (2 batches each).
// ---------------------------------------------------------------------------
__global__ void __launch_bounds__(256) sa_attn_kernel(
    const float* __restrict__ x,
    const _Float16* __restrict__ Ah, const _Float16* __restrict__ Bh,
    const float* __restrict__ u, const float* __restrict__ v,
    const float* __restrict__ c0p,
    float* __restrict__ out)
{
    __shared__ _Float16 btile[2][16 * ND];      // 2 x 4 KB double buffer

    const int lane  = threadIdx.x & 31;
    const int wave  = threadIdx.x >> 5;
    const int itile = blockIdx.x;
    const int b0    = wave * 2;                 // 8 waves x 2 batches = 16
    const int n     = lane & 15;                // tile column / A-frag row
    const int g     = lane >> 4;                // half-wave group

    const float Kc0 = K2 * (*c0p);

    // A fragments (ISA 16-bit A 16x32 layout): lane holds M=n,
    // halves 0..7 -> K = kk*32 + g*8 + t, halves 8..15 -> +16.
    v16h afrag[4];
#pragma unroll
    for (int kk = 0; kk < 4; ++kk) {
        const _Float16* pa = Ah + (itile * 16 + n) * ND + kk * 32 + g * 8;
        h8 lo = *(const h8*)(pa);
        h8 hi = *(const h8*)(pa + 16);
        afrag[kk] = __builtin_shufflevector(lo, hi, 0,1,2,3,4,5,6,7,
                                                    8,9,10,11,12,13,14,15);
    }

    // Row constants for this lane's 8 output rows (C/D layout: M = g*8 + r).
    float viK[8], xi[2][8];
#pragma unroll
    for (int r = 0; r < 8; ++r) {
        int irow = itile * 16 + g * 8 + r;
        viK[r]   = K2 * v[irow];
        xi[0][r] = x[(b0 + 0) * NN + irow];
        xi[1][r] = x[(b0 + 1) * NN + irow];
    }

    float l[16], acc[16];
#pragma unroll
    for (int t = 0; t < 16; ++t) { l[t] = 0.f; acc[t] = 0.f; }

    const float* xb0 = x + (b0 + 0) * NN;
    const float* xb1 = x + (b0 + 1) * NN;

    if (wave == 0)
        tdm_load_tile(Bh, lds_offset(&btile[0][0]));      // prologue: tile 0

    for (int jt = 0; jt < NJT; ++jt) {
        if (wave == 0)
            __builtin_amdgcn_s_wait_tensorcnt(0);          // tile jt landed
        __syncthreads();                                   // visible to all waves
        if (wave == 0 && jt + 1 < NJT)                     // stream tile jt+1
            tdm_load_tile(Bh + (size_t)(jt + 1) * 16 * ND,
                          lds_offset(&btile[(jt + 1) & 1][0]));

        const _Float16* bt = &btile[jt & 1][0];
        const int jrow = jt * 16 + n;

        // Fetch all four B fragments first (distinct registers -> one clause
        // of 8 ds_load_b128 + single s_wait_dscnt), then 4 back-to-back WMMAs.
        v16h bfrag[4];
#pragma unroll
        for (int kk = 0; kk < 4; ++kk) {
            const h8* lp = (const h8*)(bt + n * ND + kk * 32 + g * 8);
            h8 lo = lp[0];
            h8 hi = lp[2];
            bfrag[kk] = __builtin_shufflevector(lo, hi, 0,1,2,3,4,5,6,7,
                                                        8,9,10,11,12,13,14,15);
        }

        v8f c = {};
#pragma unroll
        for (int kk = 0; kk < 4; ++kk)
            c = __builtin_amdgcn_wmma_f32_16x16x32_f16(
                    false, afrag[kk], false, bfrag[kk], (short)0, c,
                    false, false);

        const float Kuj = K2 * u[jrow];
        float cK[8];
#pragma unroll
        for (int r = 0; r < 8; ++r) cK[r] = K2 * c[r];

#pragma unroll
        for (int bl = 0; bl < 2; ++bl) {
            const float xj = (bl ? xb1 : xb0)[jrow];
            const float t1 = fmaf(Kc0, xj, Kuj);           // K2*(u_j + c0*x_bj)
#pragma unroll
            for (int r = 0; r < 8; ++r) {
                const int idx = bl * 8 + r;
                float s = fmaf(xj, viK[r], cK[r]);          // K2*score (log2 dom.)
                s = fmaf(xi[bl][r], t1, s);
                float p = __builtin_amdgcn_exp2f(s);        // bare v_exp_f32
                l[idx] += p;
                acc[idx] = fmaf(p, xj, acc[idx]);
            }
        }
    }

    // Sum the 16 lanes sharing each output row (xor of lane bits 0..3 stays
    // inside the half-wave group, so row index i is invariant).
#pragma unroll
    for (int bl = 0; bl < 2; ++bl) {
#pragma unroll
        for (int r = 0; r < 8; ++r) {
            const int idx = bl * 8 + r;
            float ll = l[idx], aa = acc[idx];
#pragma unroll
            for (int off = 1; off < 16; off <<= 1) {
                ll += __shfl_xor(ll, off, 32);
                aa += __shfl_xor(aa, off, 32);
            }
            if (n == 0)
                out[(b0 + bl) * NN + itile * 16 + g * 8 + r] = aa / ll;
        }
    }
}

// ---------------------------------------------------------------------------
extern "C" void kernel_launch(void* const* d_in, const int* in_sizes, int n_in,
                              void* d_out, int out_size, void* d_ws, size_t ws_size,
                              hipStream_t stream) {
    const float* x   = (const float*)d_in[0];
    const float* q_w = (const float*)d_in[1];
    const float* q_b = (const float*)d_in[2];
    const float* k_w = (const float*)d_in[3];
    const float* k_b = (const float*)d_in[4];
    const float* pe  = (const float*)d_in[5];
    float* out = (float*)d_out;

    char* ws = (char*)d_ws;                        // needs ~2.13 MB
    _Float16* Ah = (_Float16*)ws;                  // 4096*128 f16 = 1 MB
    _Float16* Bh = (_Float16*)(ws + (size_t)NN * ND * 2);
    float* u  = (float*)(ws + 2 * (size_t)NN * ND * 2);
    float* v  = u + NN;
    float* c0 = v + NN;

    sa_prep_kernel<<<NN / 8, 256, 0, stream>>>(q_w, q_b, k_w, k_b, pe,
                                               Ah, Bh, u, v, c0);
    sa_attn_kernel<<<NN / 16, 256, 0, stream>>>(x, Ah, Bh, u, v, c0, out);
}
